// SparseAttentionFusion_23055384445752
// MI455X (gfx1250) — matrix operand order repaired and checked by reading, not compiled, not run on hardware
//
#include <hip/hip_runtime.h>
#include <hip/hip_bf16.h>
#include <math.h>

// ---------------------------------------------------------------------------
// Types & helpers
// ---------------------------------------------------------------------------
typedef __bf16 v16bf __attribute__((ext_vector_type(16)));
typedef float  v8f   __attribute__((ext_vector_type(8)));
typedef unsigned short bf16r;   // raw bf16 storage

__device__ __forceinline__ bf16r f2bf(float f) {
    unsigned u = __float_as_uint(f);
    unsigned r = (u + 0x7FFFu + ((u >> 16) & 1u)) >> 16;   // round-nearest-even
    return (bf16r)r;
}
__device__ __forceinline__ float bf2f(bf16r h) {
    return __uint_as_float(((unsigned)h) << 16);
}
__device__ __forceinline__ float gelu_exact(float x) {
    return 0.5f * x * (1.f + erff(x * 0.70710678118654752f));
}

// ---- CDNA5 async global->LDS DMA (ASYNCcnt path) via inline asm -----------
// global_load_async_to_lds_b128 vdst(LDS byte offset), vaddr(64b addr), off
// LDS operand: low 32 bits of a flat pointer to __shared__ ARE the LDS byte
// offset.
#define HAS_ASYNC_LDS 1
__device__ __forceinline__ void async_copy16(const bf16r* g, const bf16r* l) {
    unsigned lds            = (unsigned)(uintptr_t)(const void*)l;
    unsigned long long gaddr = (unsigned long long)(uintptr_t)(const void*)g;
    asm volatile("global_load_async_to_lds_b128 %0, %1, off"
                 :: "v"(lds), "v"(gaddr)
                 : "memory");
}

__device__ __forceinline__ void wait_async0() {
#if __has_builtin(__builtin_amdgcn_s_wait_asynccnt)
    __builtin_amdgcn_s_wait_asynccnt(0);
#else
    asm volatile("s_wait_asynccnt 0x0" ::: "memory");
#endif
}

#define B_    8192
#define S_    6
#define H_    1024
#define FF_   4096
#define NH_   16
#define HD_   64
#define BS_   (B_ * S_)      // 49152 rows

// GEMM tiling
#define BM 128
#define BN 128
#define BK 32
#define LDSS 40              // BK + 8 halves of pad (80B row stride, 16B aligned)

// ---------------------------------------------------------------------------
// Generic bf16 WMMA GEMM:  out[m,n] = epi( sum_k A[m,k]*Wt[n,k] + bias[n] (+resid) )
// A: [M,K] bf16 (lda = K), Wt: [N,K] bf16.  Physical out row = m*rowMul+rowAdd.
// epilogue: 0 = none, 1 = exact GELU
// ---------------------------------------------------------------------------
__global__ __launch_bounds__(256)
void gemm_bf16_wmma(const bf16r* __restrict__ A, int lda,
                    const bf16r* __restrict__ Wt,
                    const float* __restrict__ bias,
                    const float* __restrict__ resid,
                    float* __restrict__ outF,
                    bf16r* __restrict__ outH,
                    int M, int N, int K, int ldo,
                    int rowMul, int rowAdd, int epilogue)
{
    __shared__ bf16r As[2][BM * LDSS];
    __shared__ bf16r Bs[2][BN * LDSS];

    const int tid  = threadIdx.x;
    const int lane = tid & 31;           // wave32
    const int wave = tid >> 5;           // 8 waves
    const int wm   = wave >> 1;          // 0..3 -> 32-row strip
    const int wn   = wave & 1;           // 0..1 -> 64-col strip
    const int m0   = blockIdx.y * BM;
    const int n0   = blockIdx.x * BN;
    const int nk   = K / BK;

    v8f acc[2][4];
#pragma unroll
    for (int r = 0; r < 2; ++r)
#pragma unroll
        for (int c = 0; c < 4; ++c)
#pragma unroll
            for (int e = 0; e < 8; ++e) acc[r][c][e] = 0.f;

    union Frag { uint4 u[2]; v16bf v; };

    auto mma = [&](int buf) {
        Frag af[2], bfrag[4];
        // A 16x32 bf16: lane l -> M = l&15; lanes>=16 take K halves +8 / +24
        const int mrow = wm * 32 + (lane & 15);
        const int ka   = (lane >> 4) * 8;
#pragma unroll
        for (int r = 0; r < 2; ++r) {
            const bf16r* p = &As[buf][(mrow + r * 16) * LDSS + ka];
            af[r].u[0] = *reinterpret_cast<const uint4*>(p);        // K = ka..ka+7
            af[r].u[1] = *reinterpret_cast<const uint4*>(p + 16);   // K = 16+ka..+7
        }
        // B 32x16 bf16 (stored as Wt[n][k]): lane l -> N = l&15; lanes>=16 take K halves +8 / +24
        const int ncol = wn * 64 + (lane & 15);
        const int kb   = (lane >> 4) * 8;
#pragma unroll
        for (int c = 0; c < 4; ++c) {
            const bf16r* p = &Bs[buf][(ncol + c * 16) * LDSS + kb];
            bfrag[c].u[0] = *reinterpret_cast<const uint4*>(p);
            bfrag[c].u[1] = *reinterpret_cast<const uint4*>(p + 16);
        }
#pragma unroll
        for (int r = 0; r < 2; ++r)
#pragma unroll
            for (int c = 0; c < 4; ++c)
                acc[r][c] = __builtin_amdgcn_wmma_f32_16x16x32_bf16(
                    false, af[r].v, false, bfrag[c].v,
                    (short)0, acc[r][c], false, false);
    };

#if defined(HAS_ASYNC_LDS)
    // -------- async DMA pipeline: no staging VGPRs, DMA overlaps WMMA ------
    auto issue = [&](int kt, int buf) {
        const int k0 = kt * BK;
#pragma unroll
        for (int i = 0; i < 2; ++i) {
            int u = tid + i * 256;           // 512 x 16B chunks per tile
            int r = u >> 2, c = (u & 3) * 8; // 4 chunks per 32-half row
            async_copy16(A  + (size_t)(m0 + r) * lda + k0 + c, &As[buf][r * LDSS + c]);
            async_copy16(Wt + (size_t)(n0 + r) * K   + k0 + c, &Bs[buf][r * LDSS + c]);
        }
    };
    issue(0, 0);
    for (int kt = 0; kt < nk; ++kt) {
        wait_async0();            // my async writes for current buffer landed
        __syncthreads();          // everyone's landed; nobody still reads next buf
        if (kt + 1 < nk) issue(kt + 1, (kt + 1) & 1);
        mma(kt & 1);              // WMMA overlaps next tile's DMA
    }
#else
    // -------- fallback: register staging ----------------------------------
    uint4 aReg[2], bReg[2];
    auto g2r = [&](int kt) {
        const int k0 = kt * BK;
#pragma unroll
        for (int i = 0; i < 2; ++i) {
            int u = tid + i * 256;
            int r = u >> 2, c = (u & 3) * 8;
            aReg[i] = *reinterpret_cast<const uint4*>(A  + (size_t)(m0 + r) * lda + k0 + c);
            bReg[i] = *reinterpret_cast<const uint4*>(Wt + (size_t)(n0 + r) * K   + k0 + c);
        }
    };
    auto r2s = [&](int buf) {
#pragma unroll
        for (int i = 0; i < 2; ++i) {
            int u = tid + i * 256;
            int r = u >> 2, c = (u & 3) * 8;
            *reinterpret_cast<uint4*>(&As[buf][r * LDSS + c]) = aReg[i];
            *reinterpret_cast<uint4*>(&Bs[buf][r * LDSS + c]) = bReg[i];
        }
    };
    g2r(0); r2s(0);
    for (int kt = 0; kt < nk; ++kt) {
        __syncthreads();
        if (kt + 1 < nk) g2r(kt + 1);
        mma(kt & 1);
        if (kt + 1 < nk) r2s((kt + 1) & 1);
    }
#endif

    // Epilogue: C/D layout -> vgpr e holds M = e + 8*(lane>=16), N = lane&15
    const int kh = (lane >> 4) * 8;
    const int ln = lane & 15;
#pragma unroll
    for (int r = 0; r < 2; ++r) {
#pragma unroll
        for (int c = 0; c < 4; ++c) {
            const int tn = n0 + wn * 64 + c * 16 + ln;
            const float bv = bias ? bias[tn] : 0.f;
#pragma unroll
            for (int e = 0; e < 8; ++e) {
                const int tm = m0 + wm * 32 + r * 16 + kh + e;
                float v = acc[r][c][e] + bv;
                const size_t pr  = (size_t)tm * rowMul + rowAdd;
                const size_t off = pr * ldo + tn;
                if (resid) v += resid[off];
                if (epilogue == 1) v = gelu_exact(v);
                if (outF) outF[off] = v;
                if (outH) outH[off] = f2bf(v);
            }
        }
    }
}

// ---------------------------------------------------------------------------
// Prep kernels
// ---------------------------------------------------------------------------
__global__ void k_transpose_f32_bf16(const float* __restrict__ W, bf16r* __restrict__ Wt,
                                     int K, int N) {
    size_t idx = (size_t)blockIdx.x * 256 + threadIdx.x;
    if (idx >= (size_t)K * N) return;
    int k = (int)(idx / N), n = (int)(idx % N);
    Wt[(size_t)n * K + k] = f2bf(W[idx]);
}

__global__ void k_convert_f32_bf16(const float* __restrict__ src, bf16r* __restrict__ dst,
                                   size_t n) {
    size_t idx = (size_t)blockIdx.x * 256 + threadIdx.x;
    if (idx < n) dst[idx] = f2bf(src[idx]);
}

// ---------------------------------------------------------------------------
// Gate: gates[b,36] = sigmoid(proj_flat[b,6144] @ gate_W[6144,36] + gate_b)
// ---------------------------------------------------------------------------
__global__ __launch_bounds__(256)
void k_gate(const bf16r* __restrict__ projH, const float* __restrict__ gW,
            const float* __restrict__ gB, float* __restrict__ gates) {
    __shared__ float sred[36];
    const int b = blockIdx.x, tid = threadIdx.x, lane = tid & 31;
    if (tid < 36) sred[tid] = 0.f;
    __syncthreads();
    float pacc[36];
#pragma unroll
    for (int n = 0; n < 36; ++n) pacc[n] = 0.f;
    const bf16r* row = projH + (size_t)b * (S_ * H_);
    for (int k = tid; k < S_ * H_; k += 256) {
        const float a = bf2f(row[k]);
        const float* wr = gW + (size_t)k * 36;
#pragma unroll
        for (int n = 0; n < 36; ++n) pacc[n] += a * wr[n];
    }
    // wave32 shuffle reduction, then one atomic per wave
#pragma unroll
    for (int n = 0; n < 36; ++n) {
#pragma unroll
        for (int off = 16; off > 0; off >>= 1)
            pacc[n] += __shfl_down(pacc[n], off, 32);
    }
    if (lane == 0) {
#pragma unroll
        for (int n = 0; n < 36; ++n) atomicAdd(&sred[n], pacc[n]);
    }
    __syncthreads();
    if (tid < 36) {
        const float s = sred[tid] + gB[tid];
        gates[(size_t)b * 36 + tid] = 1.f / (1.f + expf(-s));
    }
}

// gated[b,i,h] = sum_j gates[b,i,j] * values[b,j,h]
__global__ void k_gating_combine(const float* __restrict__ gates,
                                 const bf16r* __restrict__ valH,
                                 bf16r* __restrict__ gatedH) {
    const size_t idx = (size_t)blockIdx.x * 256 + threadIdx.x;   // B*H threads
    const int b = (int)(idx >> 10), h = (int)(idx & 1023);
    float v[S_];
#pragma unroll
    for (int j = 0; j < S_; ++j) v[j] = bf2f(valH[((size_t)b * S_ + j) * H_ + h]);
    const float* g = gates + (size_t)b * 36;
#pragma unroll
    for (int i = 0; i < S_; ++i) {
        float a = 0.f;
#pragma unroll
        for (int j = 0; j < S_; ++j) a += g[i * S_ + j] * v[j];
        gatedH[((size_t)b * S_ + i) * H_ + h] = f2bf(a);
    }
}

// ---------------------------------------------------------------------------
// LayerNorm over H=1024 (one block / row); in-place f32 allowed
// ---------------------------------------------------------------------------
__global__ __launch_bounds__(256)
void k_layernorm(const float* __restrict__ x, const float* __restrict__ g,
                 const float* __restrict__ bta, float* __restrict__ outF,
                 bf16r* __restrict__ outH) {
    __shared__ float s1[256], s2[256];
    const int row = blockIdx.x, tid = threadIdx.x;
    const float* xr = x + (size_t)row * H_;
    float v[4], sum = 0.f, sq = 0.f;
#pragma unroll
    for (int i = 0; i < 4; ++i) {
        v[i] = xr[tid + i * 256];
        sum += v[i]; sq += v[i] * v[i];
    }
    s1[tid] = sum; s2[tid] = sq;
    __syncthreads();
    for (int off = 128; off > 0; off >>= 1) {
        if (tid < off) { s1[tid] += s1[tid + off]; s2[tid] += s2[tid + off]; }
        __syncthreads();
    }
    const float mean = s1[0] * (1.f / H_);
    const float var  = s2[0] * (1.f / H_) - mean * mean;
    const float rstd = rsqrtf(var + 1e-5f);
#pragma unroll
    for (int i = 0; i < 4; ++i) {
        const int h = tid + i * 256;
        const float o = (v[i] - mean) * rstd * g[h] + bta[h];
        if (outF) outF[(size_t)row * H_ + h] = o;
        if (outH) outH[(size_t)row * H_ + h] = f2bf(o);
    }
}

// ---------------------------------------------------------------------------
// 6-token, 16-head attention core: one block per (batch, head), 64 threads
// ---------------------------------------------------------------------------
__global__ __launch_bounds__(64)
void k_attention(const bf16r* __restrict__ Q, const bf16r* __restrict__ K,
                 const bf16r* __restrict__ V, bf16r* __restrict__ O) {
    __shared__ float Qs[S_ * HD_], Ks[S_ * HD_], Vs[S_ * HD_], P[S_ * S_];
    const int b = blockIdx.x >> 4, head = blockIdx.x & 15, t = threadIdx.x;
    const size_t base = (size_t)b * S_ * H_ + head * HD_ + t;
#pragma unroll
    for (int s = 0; s < S_; ++s) {
        Qs[s * HD_ + t] = bf2f(Q[base + (size_t)s * H_]);
        Ks[s * HD_ + t] = bf2f(K[base + (size_t)s * H_]);
        Vs[s * HD_ + t] = bf2f(V[base + (size_t)s * H_]);
    }
    __syncthreads();
    if (t < S_ * S_) {
        const int q = t / S_, k = t % S_;
        float d = 0.f;
#pragma unroll
        for (int e = 0; e < HD_; ++e) d += Qs[q * HD_ + e] * Ks[k * HD_ + e];
        P[t] = d * 0.125f;   // 1/sqrt(64)
    }
    __syncthreads();
    if (t < S_) {
        float mx = -1e30f;
#pragma unroll
        for (int k = 0; k < S_; ++k) mx = fmaxf(mx, P[t * S_ + k]);
        float e[S_], s = 0.f;
#pragma unroll
        for (int k = 0; k < S_; ++k) { e[k] = expf(P[t * S_ + k] - mx); s += e[k]; }
        const float inv = 1.f / s;
#pragma unroll
        for (int k = 0; k < S_; ++k) P[t * S_ + k] = e[k] * inv;
    }
    __syncthreads();
#pragma unroll
    for (int q = 0; q < S_; ++q) {
        float o = 0.f;
#pragma unroll
        for (int k = 0; k < S_; ++k) o += P[q * S_ + k] * Vs[k * HD_ + t];
        O[base + (size_t)q * H_] = f2bf(o);
    }
}

// out[b,h] = mean_s x[b,s,h]
__global__ void k_token_mean(const float* __restrict__ x, float* __restrict__ out) {
    const size_t idx = (size_t)blockIdx.x * 256 + threadIdx.x;   // B*H
    const int b = (int)(idx >> 10), h = (int)(idx & 1023);
    float s = 0.f;
#pragma unroll
    for (int t = 0; t < S_; ++t) s += x[((size_t)b * S_ + t) * H_ + h];
    out[(size_t)b * H_ + h] = s * (1.f / S_);
}

// ---------------------------------------------------------------------------
// Host-side orchestration
// ---------------------------------------------------------------------------
enum {
    IN_S0 = 0,                  // 6 streams
    IN_STREAM_W0 = 6,           // stream_W[0..5]
    IN_STREAM_B0 = 12,          // stream_b[0..5]
    IN_GATE_W = 18, IN_GATE_B = 19,
    IN_VAL_W = 20, IN_VAL_B, IN_GOUT_W, IN_GOUT_B,
    IN_Q_W, IN_Q_B, IN_K_W, IN_K_B, IN_V_W, IN_V_B, IN_O_W, IN_O_B,
    IN_FF1_W, IN_FF1_B, IN_FF2_W, IN_FF2_B,
    IN_GNORM_G, IN_GNORM_B, IN_N1_G, IN_N1_B, IN_N2_G, IN_N2_B
};

static const int STREAM_DIMS[S_] = {512, 512, 256, 256, 128, 128};

extern "C" void kernel_launch(void* const* d_in, const int* in_sizes, int n_in,
                              void* d_out, int out_size, void* d_ws, size_t ws_size,
                              hipStream_t stream) {
    (void)in_sizes; (void)n_in; (void)out_size; (void)ws_size;
    auto FIN = [&](int i) { return (const float*)d_in[i]; };

    // ---- workspace bump allocator (256B aligned) ----
    char* base = (char*)d_ws;
    size_t off = 0;
    auto alloc = [&](size_t bytes) -> char* {
        char* p = base + off;
        off += (bytes + 255) & ~(size_t)255;
        return p;
    };

    bf16r* wst[S_];
    for (int s = 0; s < S_; ++s) wst[s] = (bf16r*)alloc((size_t)STREAM_DIMS[s] * H_ * 2);
    bf16r* wval  = (bf16r*)alloc((size_t)H_ * H_ * 2);
    bf16r* wgout = (bf16r*)alloc((size_t)H_ * H_ * 2);
    bf16r* wq    = (bf16r*)alloc((size_t)H_ * H_ * 2);
    bf16r* wk    = (bf16r*)alloc((size_t)H_ * H_ * 2);
    bf16r* wv    = (bf16r*)alloc((size_t)H_ * H_ * 2);
    bf16r* wo    = (bf16r*)alloc((size_t)H_ * H_ * 2);
    bf16r* wff1  = (bf16r*)alloc((size_t)H_ * FF_ * 2);  // [FF][H]
    bf16r* wff2  = (bf16r*)alloc((size_t)FF_ * H_ * 2);  // [H][FF]
    bf16r* sbh[S_];
    for (int s = 0; s < S_; ++s) sbh[s] = (bf16r*)alloc((size_t)B_ * STREAM_DIMS[s] * 2);
    float* projF = (float*)alloc((size_t)BS_ * H_ * 4);
    bf16r* R1 = (bf16r*)alloc((size_t)BS_ * H_ * 2);   // proj_bf16 -> q_bf16 -> h_bf16
    bf16r* R2 = (bf16r*)alloc((size_t)BS_ * H_ * 2);   // values    -> k_bf16
    bf16r* R3 = (bf16r*)alloc((size_t)BS_ * H_ * 2);   // gated     -> v_bf16
    bf16r* R4 = (bf16r*)alloc((size_t)BS_ * H_ * 2);   // xn        -> attnout
    float* gates = (float*)alloc((size_t)B_ * 36 * 4);
    float* xF    = (float*)alloc((size_t)BS_ * H_ * 4);
    bf16r* ffmid = (bf16r*)alloc((size_t)BS_ * FF_ * 2);

    auto tpose = [&](const float* W, bf16r* Wt, int K, int N) {
        size_t tot = (size_t)K * N;
        k_transpose_f32_bf16<<<(unsigned)((tot + 255) / 256), 256, 0, stream>>>(W, Wt, K, N);
    };
    auto cvt = [&](const float* src, bf16r* dst, size_t n) {
        k_convert_f32_bf16<<<(unsigned)((n + 255) / 256), 256, 0, stream>>>(src, dst, n);
    };
    auto gemm = [&](const bf16r* A, int lda, const bf16r* Wt, const float* bias,
                    const float* resid, float* oF, bf16r* oH,
                    int M, int N, int K, int ldo, int rowMul, int rowAdd, int epi) {
        dim3 grid(N / BN, M / BM);
        gemm_bf16_wmma<<<grid, 256, 0, stream>>>(A, lda, Wt, bias, resid, oF, oH,
                                                 M, N, K, ldo, rowMul, rowAdd, epi);
    };

    // ---- weight prep (bf16, transposed to [N][K]) ----
    for (int s = 0; s < S_; ++s) tpose(FIN(IN_STREAM_W0 + s), wst[s], STREAM_DIMS[s], H_);
    tpose(FIN(IN_VAL_W),  wval,  H_, H_);
    tpose(FIN(IN_GOUT_W), wgout, H_, H_);
    tpose(FIN(IN_Q_W),    wq,    H_, H_);
    tpose(FIN(IN_K_W),    wk,    H_, H_);
    tpose(FIN(IN_V_W),    wv,    H_, H_);
    tpose(FIN(IN_O_W),    wo,    H_, H_);
    tpose(FIN(IN_FF1_W),  wff1,  H_, FF_);
    tpose(FIN(IN_FF2_W),  wff2,  FF_, H_);
    for (int s = 0; s < S_; ++s) cvt(FIN(IN_S0 + s), sbh[s], (size_t)B_ * STREAM_DIMS[s]);

    // ---- stream projections -> proj [B,6,H] (f32 + bf16) ----
    for (int s = 0; s < S_; ++s)
        gemm(sbh[s], STREAM_DIMS[s], wst[s], FIN(IN_STREAM_B0 + s), nullptr,
             projF, R1, B_, H_, STREAM_DIMS[s], H_, S_, s, 0);

    // ---- cross-stream gating ----
    k_gate<<<B_, 256, 0, stream>>>(R1, FIN(IN_GATE_W), FIN(IN_GATE_B), gates);
    gemm(R1, H_, wval, FIN(IN_VAL_B), nullptr, nullptr, R2, BS_, H_, H_, H_, 1, 0, 0);
    k_gating_combine<<<(B_ * H_) / 256, 256, 0, stream>>>(gates, R2, R3);
    gemm(R3, H_, wgout, FIN(IN_GOUT_B), projF, xF, nullptr, BS_, H_, H_, H_, 1, 0, 0);
    k_layernorm<<<BS_, 256, 0, stream>>>(xF, FIN(IN_GNORM_G), FIN(IN_GNORM_B), xF, nullptr);

    // ---- tiny self-attention ----
    k_layernorm<<<BS_, 256, 0, stream>>>(xF, FIN(IN_N1_G), FIN(IN_N1_B), nullptr, R4);
    gemm(R4, H_, wq, FIN(IN_Q_B), nullptr, nullptr, R1, BS_, H_, H_, H_, 1, 0, 0);
    gemm(R4, H_, wk, FIN(IN_K_B), nullptr, nullptr, R2, BS_, H_, H_, H_, 1, 0, 0);
    gemm(R4, H_, wv, FIN(IN_V_B), nullptr, nullptr, R3, BS_, H_, H_, H_, 1, 0, 0);
    k_attention<<<B_ * NH_, 64, 0, stream>>>(R1, R2, R3, R4);
    gemm(R4, H_, wo, FIN(IN_O_B), xF, xF, nullptr, BS_, H_, H_, H_, 1, 0, 0);

    // ---- FFN ----
    k_layernorm<<<BS_, 256, 0, stream>>>(xF, FIN(IN_N2_G), FIN(IN_N2_B), nullptr, R1);
    gemm(R1, H_, wff1, FIN(IN_FF1_B), nullptr, nullptr, ffmid, BS_, FF_, H_, FF_, 1, 0, 1);
    gemm(ffmid, FF_, wff2, FIN(IN_FF2_B), xF, xF, nullptr, BS_, H_, FF_, H_, 1, 0, 0);

    // ---- token mean -> output ----
    k_token_mean<<<(B_ * H_) / 256, 256, 0, stream>>>(xF, (float*)d_out);
}
